// Attend_5025111736438
// MI455X (gfx1250) — matrix operand order repaired and checked by reading, not compiled, not run on hardware
//
#include <hip/hip_runtime.h>

// Attention: B=2, H=16, S=2048, D=64, fp32 in/out.
// Flash-attention single pass, f16 WMMA (v_wmma_f32_16x16x32_f16) w/ f32 accum.
// Row-sum via P x ones WMMA; row-max via DPP16 row_ror butterfly (no LDS).
// BK=128 key blocks; scalar-base addressing; software-pipelined LDS fragment loads.

typedef _Float16 v16h __attribute__((ext_vector_type(16)));
typedef _Float16 v8h  __attribute__((ext_vector_type(8)));
typedef _Float16 v4h  __attribute__((ext_vector_type(4)));
typedef float    v8f  __attribute__((ext_vector_type(8)));

namespace {
constexpr int Bc = 2;
constexpr int Hc = 16;
constexpr int Sc = 2048;
constexpr int Dc = 64;
constexpr int BM = 128;       // Q rows per workgroup (8 waves x 16)
constexpr int BK = 128;       // keys per iteration
constexpr int NT = BK / 16;   // 16-key subtiles per block (8)
constexpr int NC = BK / 32;   // 32-key WMMA chunks per block (4)
}

// DPP16 row_ror rotation within each 16-lane row (VALU, no DS traffic).
template <int CTRL>
__device__ __forceinline__ float dpp_ror_f32(float x) {
  const int i = __builtin_bit_cast(int, x);
  const int r = __builtin_amdgcn_update_dpp(i, i, CTRL, 0xf, 0xf, true);
  return __builtin_bit_cast(float, r);
}

// max across the 16 lanes of a half-row via ror 1/2/4/8 butterfly
__device__ __forceinline__ float rowmax16(float x) {
  x = fmaxf(x, dpp_ror_f32<0x121>(x));   // row_ror:1
  x = fmaxf(x, dpp_ror_f32<0x122>(x));   // row_ror:2
  x = fmaxf(x, dpp_ror_f32<0x124>(x));   // row_ror:4
  x = fmaxf(x, dpp_ror_f32<0x128>(x));   // row_ror:8
  return x;
}

__global__ __launch_bounds__(256) void fa_fwd_kernel(
    const float* __restrict__ q, const float* __restrict__ k,
    const float* __restrict__ v, float* __restrict__ o) {
  __shared__ _Float16 Klds[BK * Dc];       // [key][d]   row stride 64h = 128B
  __shared__ _Float16 Vlds[Dc * BK];       // [d][key]   transposed V tile
  __shared__ _Float16 Plds[8 * 16 * BK];   // per-wave P staging [wave][m][key]

  const int tid   = threadIdx.x;
  const int wave  = tid >> 5;
  const int lane  = tid & 31;
  const int half_ = lane >> 4;   // 0 | 1 (wave32 halves)
  const int l16   = lane & 15;

  const int bh   = blockIdx.y;                    // fused batch*head, 0..31
  const int row0 = blockIdx.x * BM + wave * 16;   // this wave's Q tile base row

  const float* qb = q + (size_t)bh * Sc * Dc;
  const float* kb = k + (size_t)bh * Sc * Dc;
  const float* vb = v + (size_t)bh * Sc * Dc;
  float*       ob = o + (size_t)bh * Sc * Dc;

  // fold 1/sqrt(D) and log2(e) into Q so softmax runs in base-2 (v_exp_f32)
  const float qscale = 0.125f * 1.44269504088896340736f;

  // ---- Q tile -> A-matrix fragments (16x32 f16, contiguous k order: d = 32c + 16*half + h) ----
  v16h afrag[2];
  {
    const float* qrow = qb + (size_t)(row0 + l16) * Dc;
#pragma unroll
    for (int c = 0; c < 2; ++c) {
#pragma unroll
      for (int h = 0; h < 16; ++h) {
        const int d = c * 32 + h + half_ * 16;
        afrag[c][h] = (_Float16)(qrow[d] * qscale);
      }
    }
  }

  // constant all-ones B fragment: rowsum(P) = P x ones via WMMA
  v16h onesb;
#pragma unroll
  for (int i = 0; i < 16; ++i) onesb[i] = (_Float16)1.0f;

  // per-row online-softmax state; row m = r + 8*laneHalf
  float mrow[8];
  v8f acc[4];   // O accumulator, 4 d-tiles
  v8f accL;     // softmax denominator accumulator (P x ones)
#pragma unroll
  for (int r = 0; r < 8; ++r) { mrow[r] = -__builtin_inff(); accL[r] = 0.0f; }
#pragma unroll
  for (int t = 0; t < 4; ++t) {
#pragma unroll
    for (int r = 0; r < 8; ++r) acc[t][r] = 0.0f;
  }

  _Float16* Pw = &Plds[wave * 16 * BK];

  // thread-constant tile offsets (one pointer per array; per-i offsets become
  // compile-time immediates since i*256 float4s == i*16 full rows)
  const int key0 = tid >> 4;            // 0..15
  const int d4   = (tid & 15) << 2;     // 0..60
  const int goff = key0 * Dc + d4;      // element offset into a tile
  _Float16* const kld0 = &Klds[goff];
  _Float16* const vld0 = &Vlds[d4 * BK + key0];

#pragma unroll 1
  for (int kb0 = 0; kb0 < Sc; kb0 += BK) {
    __syncthreads();   // previous block's LDS consumers done

    const float* kp = kb + (size_t)kb0 * Dc + goff;
    const float* vp = vb + (size_t)kb0 * Dc + goff;

    // ---- cooperative K/V tile load + f32->f16 convert (8192 elems each) ----
#pragma unroll
    for (int i = 0; i < 8; ++i) {
      const float4 k4 = *(const float4*)(kp + i * 16 * Dc);
      const float4 v4 = *(const float4*)(vp + i * 16 * Dc);
      const v4h kk = { (_Float16)k4.x, (_Float16)k4.y,
                       (_Float16)k4.z, (_Float16)k4.w };
      *(v4h*)(kld0 + i * 16 * Dc) = kk;          // one ds_store_b64
      _Float16* vl = vld0 + i * 16;              // transposed: scatter 4 x b16
      vl[0 * BK] = (_Float16)v4.x;
      vl[1 * BK] = (_Float16)v4.y;
      vl[2 * BK] = (_Float16)v4.z;
      vl[3 * BK] = (_Float16)v4.w;
    }
    __syncthreads();

    // prefetch next K/V block into cache (global_prefetch_b8); one address/thread
    // x 256 threads spans only the first 16 keys (4KB) of each array's next tile
    if (kb0 + BK < Sc) {
      __builtin_prefetch(kp + BK * Dc, 0, 1);
      __builtin_prefetch(vp + BK * Dc, 0, 1);
    }

    // ---- S = Q*K^T : 8 subtiles, 2-deep software-pipelined B fragments ----
    v8f st[NT];
    {
      v16h b0 = *(const v16h*)&Klds[(0 * 16 + l16) * Dc + 0 * 32 + half_ * 16];
      v16h b1 = *(const v16h*)&Klds[(0 * 16 + l16) * Dc + 1 * 32 + half_ * 16];
#pragma unroll
      for (int t = 0; t < NT; ++t) {
        v16h n0 = b0, n1 = b1;
        if (t + 1 < NT) {
          n0 = *(const v16h*)&Klds[((t + 1) * 16 + l16) * Dc + 0 * 32 + half_ * 16];
          n1 = *(const v16h*)&Klds[((t + 1) * 16 + l16) * Dc + 1 * 32 + half_ * 16];
        }
        v8f c8 = {};
        c8 = __builtin_amdgcn_wmma_f32_16x16x32_f16(
            false, afrag[0], false, b0, (short)0, c8, false, false);
        c8 = __builtin_amdgcn_wmma_f32_16x16x32_f16(
            false, afrag[1], false, b1, (short)0, c8, false, false);
        st[t] = c8;
        b0 = n0; b1 = n1;
      }
    }

    // ---- online softmax (base-2): row-max via DPP butterfly, pure VALU ----
#pragma unroll
    for (int r = 0; r < 8; ++r) {
      float bm = fmaxf(st[0][r], st[1][r]);
      bm = fmaxf(bm, fmaxf(st[2][r], st[3][r]));
      bm = fmaxf(bm, fmaxf(st[4][r], st[5][r]));
      bm = fmaxf(bm, fmaxf(st[6][r], st[7][r]));
      bm = rowmax16(bm);
      const float mnew  = fmaxf(mrow[r], bm);
      const float alpha = __builtin_amdgcn_exp2f(mrow[r] - mnew);
      mrow[r] = mnew;
      accL[r] *= alpha;
#pragma unroll
      for (int t = 0; t < 4; ++t) acc[t][r] *= alpha;
#pragma unroll
      for (int t = 0; t < NT; ++t)
        st[t][r] = __builtin_amdgcn_exp2f(st[t][r] - mnew);
    }

    // ---- P: C layout -> [m][key] f16 staging (wave-private, in-order DS) ----
#pragma unroll
    for (int t = 0; t < NT; ++t) {
#pragma unroll
      for (int r = 0; r < 8; ++r)
        Pw[(r + half_ * 8) * BK + t * 16 + l16] = (_Float16)st[t][r];
    }

    // ---- O += P * V and accL += P * ones ----
    // per chunk: batch-load all 4 V fragments (8 parallel ds_load_b128),
    // then issue 5 WMMAs against one P A-fragment
#pragma unroll
    for (int c = 0; c < NC; ++c) {
      v16h vbf[4];
#pragma unroll
      for (int t = 0; t < 4; ++t)
        vbf[t] = *(const v16h*)&Vlds[(t * 16 + l16) * BK + c * 32 + half_ * 16];

      const _Float16* prow = &Pw[l16 * BK + c * 32 + half_ * 16];
      const v8h lo = *(const v8h*)(prow);
      const v8h hi = *(const v8h*)(prow + 8);
      v16h pa;
#pragma unroll
      for (int i = 0; i < 8; ++i) { pa[i] = lo[i]; pa[i + 8] = hi[i]; }

      // softmax denominator: rowsum of this key chunk, replicated over lanes
      accL = __builtin_amdgcn_wmma_f32_16x16x32_f16(
          false, pa, false, onesb, (short)0, accL, false, false);

#pragma unroll
      for (int t = 0; t < 4; ++t)
        acc[t] = __builtin_amdgcn_wmma_f32_16x16x32_f16(
            false, pa, false, vbf[t], (short)0, acc[t], false, false);
    }
  }

  // ---- epilogue: O /= l, store fp32 ----
#pragma unroll
  for (int r = 0; r < 8; ++r) {
    const float inv = 1.0f / accL[r];
    const int row = row0 + r + half_ * 8;
#pragma unroll
    for (int t = 0; t < 4; ++t) {
      const int d = t * 16 + l16;
      ob[(size_t)row * Dc + d] = acc[t][r] * inv;
    }
  }
}

extern "C" void kernel_launch(void* const* d_in, const int* in_sizes, int n_in,
                              void* d_out, int out_size, void* d_ws, size_t ws_size,
                              hipStream_t stream) {
  const float* q = (const float*)d_in[0];
  const float* k = (const float*)d_in[1];
  const float* v = (const float*)d_in[2];
  float* o = (float*)d_out;
  dim3 grid(Sc / BM, Bc * Hc);   // (16, 32)
  fa_fwd_kernel<<<grid, dim3(256), 0, stream>>>(q, k, v, o);
}
